// LSTMModel_73065983640196
// MI455X (gfx1250) — compile-verified, matrix operand order checked by reading, not run on hardware
//
#include <hip/hip_runtime.h>
#include <hip/hip_bf16.h>

typedef __attribute__((ext_vector_type(16))) __bf16 v16bf;
typedef __attribute__((ext_vector_type(8)))  float  v8f;

// ---------------------------------------------------------------------------
// Fragment loader: 16-bit A/B matrix layout for v_wmma_*_16x16x32_bf16.
// Lane l (wave32): row/col = l&15, kbase = (l>>4)*8.
// Vector elements 0..7  hold K = kbase..kbase+7       (contiguous 16 bytes)
// Vector elements 8..15 hold K = 16+kbase..16+kbase+7 (contiguous 16 bytes)
// `p` must already point at element (k0 + kbase) of the row.
// ---------------------------------------------------------------------------
__device__ __forceinline__ v16bf load_frag(const __hip_bfloat16* p) {
    union { uint4 u[2]; v16bf v; } t;
    t.u[0] = *reinterpret_cast<const uint4*>(p);
    t.u[1] = *reinterpret_cast<const uint4*>(p + 16);
    return t.v;
}

__device__ __forceinline__ float sigmf(float x) {
    return 1.0f / (1.0f + __expf(-x));
}

// ---------------------------------------------------------------------------
// One LSTM timestep body:  gates = [hprev, x_t] @ [Wh, Wx]^T + b_ih + b_hh
// 64 blocks per layer-step; block `bl` owns 16 hidden units (all 4 gates).
// block = 256 threads = 8 waves.  Tile grid: 4 M-tiles (batch 64) x 4 N-tiles
// (4 gates x 16 cols).  Wave w: n-tile = w&3 (== gate), m-tiles {2*(w>>2),
// 2*(w>>2)+1}  ->  2 WMMA accumulators per wave, B-fragment reused 2x.
// ---------------------------------------------------------------------------
__device__ __forceinline__
void lstm_step_body(int bl,
                    const __hip_bfloat16* __restrict__ hprev,   // [64,1024]
                    const __hip_bfloat16* __restrict__ xt,      // rows, stride xstride
                    int Kx, int xstride,
                    const __hip_bfloat16* __restrict__ Wh,      // [4096,1024]
                    const __hip_bfloat16* __restrict__ Wx,      // [4096,Kx]
                    const float* __restrict__ b_ih,
                    const float* __restrict__ b_hh,
                    float* __restrict__ c,                      // [64,1024]
                    __hip_bfloat16* __restrict__ hnext,         // [64,1024]
                    float gs[64][64])
{
    const int tid   = threadIdx.x;
    const int wave  = tid >> 5;
    const int lane  = tid & 31;
    const int mlane = lane & 15;
    const int kb    = (lane >> 4) * 8;
    const int j0    = bl * 16;

    const int nt    = wave & 3;              // gate index (i,f,g,o)
    const int mh    = wave >> 2;             // m-tiles {2mh, 2mh+1}
    const int nglob = nt * 1024 + j0 + mlane;

    v8f acc[2];
    v8f zero = {0.f, 0.f, 0.f, 0.f, 0.f, 0.f, 0.f, 0.f};
    acc[0] = zero; acc[1] = zero;

    // K over recurrent input h_prev (K = 1024)
    const __hip_bfloat16* wrow_h = Wh + (size_t)nglob * 1024;
    for (int k0 = 0; k0 < 1024; k0 += 32) {
        v16bf bfrag = load_frag(wrow_h + k0 + kb);
#pragma unroll
        for (int i = 0; i < 2; ++i) {
            int mt = mh * 2 + i;
            v16bf afrag = load_frag(hprev + (size_t)(mt * 16 + mlane) * 1024 + k0 + kb);
            acc[i] = __builtin_amdgcn_wmma_f32_16x16x32_bf16(
                false, afrag, false, bfrag, (short)0, acc[i], false, false);
        }
    }
    // K over layer input x_t (K = Kx), fused (no precomputed xg buffer)
    const __hip_bfloat16* wrow_x = Wx + (size_t)nglob * Kx;
    for (int k0 = 0; k0 < Kx; k0 += 32) {
        v16bf bfrag = load_frag(wrow_x + k0 + kb);
#pragma unroll
        for (int i = 0; i < 2; ++i) {
            int mt = mh * 2 + i;
            v16bf afrag = load_frag(xt + (size_t)(mt * 16 + mlane) * xstride + k0 + kb);
            acc[i] = __builtin_amdgcn_wmma_f32_16x16x32_bf16(
                false, afrag, false, bfrag, (short)0, acc[i], false, false);
        }
    }

    // C/D layout: VGPR v holds M = v + 8*(lane>>4), N = lane&15.
    const float bias = b_ih[nglob] + b_hh[nglob];
    const int q = nt * 16 + mlane;           // LDS col: [0..15]=i [16..31]=f [32..47]=g [48..63]=o
#pragma unroll
    for (int i = 0; i < 2; ++i) {
        int mt = mh * 2 + i;
#pragma unroll
        for (int v = 0; v < 8; ++v)
            gs[mt * 16 + kb + v][q] = acc[i][v] + bias;
    }

    __syncthreads();

    // Elementwise cell update: 64 batch x 16 hidden units = 1024 elems, 4/thread
#pragma unroll
    for (int r = 0; r < 4; ++r) {
        int e  = tid + r * 256;
        int b  = e >> 4;
        int cc = e & 15;
        float iv = gs[b][cc];
        float fv = gs[b][16 + cc];
        float gv = gs[b][32 + cc];
        float ov = gs[b][48 + cc];
        int idx = b * 1024 + j0 + cc;
        float cp = c[idx];
        float cn = sigmf(fv) * cp + sigmf(iv) * tanhf(gv);
        float hn = sigmf(ov) * tanhf(cn);
        c[idx] = cn;
        hnext[idx] = __float2bfloat16(hn);
    }
}

// ---------------------------------------------------------------------------
// Dual-layer pipelined step: blocks 0..63 run layer-0 step t, blocks 64..127
// run layer-1 step t-1 (its inputs were finished by the previous launch).
// Halves the number of launches on the sequential critical path and doubles
// the number of busy WGPs per launch.
// ---------------------------------------------------------------------------
__global__ __launch_bounds__(256)
void lstm_dual_step(// layer 0 step
                    const __hip_bfloat16* __restrict__ hprev0,
                    const __hip_bfloat16* __restrict__ xt0, int Kx0, int xstride0,
                    const __hip_bfloat16* __restrict__ Wh0,
                    const __hip_bfloat16* __restrict__ Wx0,
                    const float* __restrict__ bih0, const float* __restrict__ bhh0,
                    float* __restrict__ c0, __hip_bfloat16* __restrict__ hnext0,
                    int do0,
                    // layer 1 step
                    const __hip_bfloat16* __restrict__ hprev1,
                    const __hip_bfloat16* __restrict__ xt1, int Kx1, int xstride1,
                    const __hip_bfloat16* __restrict__ Wh1,
                    const __hip_bfloat16* __restrict__ Wx1,
                    const float* __restrict__ bih1, const float* __restrict__ bhh1,
                    float* __restrict__ c1, __hip_bfloat16* __restrict__ hnext1,
                    int do1)
{
    __shared__ float gs[64][64];   // 16 KB gate staging: [batch][4 gates * 16]
    const int bl = blockIdx.x;
    if (bl < 64) {
        if (do0)
            lstm_step_body(bl, hprev0, xt0, Kx0, xstride0, Wh0, Wx0,
                           bih0, bhh0, c0, hnext0, gs);
    } else {
        if (do1)
            lstm_step_body(bl - 64, hprev1, xt1, Kx1, xstride1, Wh1, Wx1,
                           bih1, bhh1, c1, hnext1, gs);
    }
}

// ---------------------------------------------------------------------------
// FC on last timestep: out[64,256] = h[64,1024] @ Wfc^T + b_fc   (fp32 out)
// grid = 4 blocks x 64 output cols; wave w: n-tile = w&3, m-half = w>>2
// ---------------------------------------------------------------------------
__global__ __launch_bounds__(256)
void fc_kernel(const __hip_bfloat16* __restrict__ h,   // [64,1024]
               const __hip_bfloat16* __restrict__ W,   // [256,1024]
               const float* __restrict__ bias,
               float* __restrict__ out)                // [64,256]
{
    const int tid   = threadIdx.x;
    const int wave  = tid >> 5;
    const int lane  = tid & 31;
    const int mlane = lane & 15;
    const int kb    = (lane >> 4) * 8;

    const int nt    = wave & 3;
    const int mh    = wave >> 2;            // m-tiles {0,1} or {2,3}
    const int nglob = blockIdx.x * 64 + nt * 16 + mlane;

    v8f acc[2];
    v8f zero = {0.f, 0.f, 0.f, 0.f, 0.f, 0.f, 0.f, 0.f};
    acc[0] = zero; acc[1] = zero;

    const __hip_bfloat16* wrow = W + (size_t)nglob * 1024;
    for (int k0 = 0; k0 < 1024; k0 += 32) {
        v16bf bfrag = load_frag(wrow + k0 + kb);
#pragma unroll
        for (int i = 0; i < 2; ++i) {
            int mt = mh * 2 + i;
            v16bf afrag = load_frag(h + (size_t)(mt * 16 + mlane) * 1024 + k0 + kb);
            acc[i] = __builtin_amdgcn_wmma_f32_16x16x32_bf16(
                false, afrag, false, bfrag, (short)0, acc[i], false, false);
        }
    }

    float bv = bias[nglob];
#pragma unroll
    for (int i = 0; i < 2; ++i) {
        int mt = mh * 2 + i;
#pragma unroll
        for (int v = 0; v < 8; ++v) {
            int m = mt * 16 + kb + v;
            out[m * 256 + nglob] = acc[i][v] + bv;
        }
    }
}

// ---------------------------------------------------------------------------
// Helpers: fp32 -> bf16 conversion, zero fills (graph-capture safe)
// ---------------------------------------------------------------------------
__global__ void cvt_f32_bf16(const float* __restrict__ in,
                             __hip_bfloat16* __restrict__ out, int n) {
    int i = blockIdx.x * blockDim.x + threadIdx.x;
    if (i < n) out[i] = __float2bfloat16(in[i]);
}
__global__ void fill_zero_bf16(__hip_bfloat16* p, int n) {
    int i = blockIdx.x * blockDim.x + threadIdx.x;
    if (i < n) p[i] = __float2bfloat16(0.0f);
}
__global__ void fill_zero_f32(float* p, int n) {
    int i = blockIdx.x * blockDim.x + threadIdx.x;
    if (i < n) p[i] = 0.0f;
}

// ---------------------------------------------------------------------------
extern "C" void kernel_launch(void* const* d_in, const int* in_sizes, int n_in,
                              void* d_out, int out_size, void* d_ws, size_t ws_size,
                              hipStream_t stream) {
    (void)in_sizes; (void)n_in; (void)out_size; (void)ws_size;
    const float* x    = (const float*)d_in[0];
    const float* wih0 = (const float*)d_in[1];
    const float* whh0 = (const float*)d_in[2];
    const float* bih0 = (const float*)d_in[3];
    const float* bhh0 = (const float*)d_in[4];
    const float* wih1 = (const float*)d_in[5];
    const float* whh1 = (const float*)d_in[6];
    const float* bih1 = (const float*)d_in[7];
    const float* bhh1 = (const float*)d_in[8];
    const float* wfc  = (const float*)d_in[9];
    const float* bfc  = (const float*)d_in[10];
    float* out = (float*)d_out;

    const int B = 64, T = 512, IN = 256, H = 1024, OUT = 256;

    char* base = (char*)d_ws;
    size_t off = 0;
    auto alloc = [&](size_t bytes) -> void* {
        off = (off + 255) & ~(size_t)255;
        void* p = base + off;
        off += bytes;
        return p;
    };
    __hip_bfloat16* xbf   = (__hip_bfloat16*)alloc((size_t)B * T * IN * 2);
    __hip_bfloat16* whh0b = (__hip_bfloat16*)alloc((size_t)4 * H * H * 2);
    __hip_bfloat16* wih0b = (__hip_bfloat16*)alloc((size_t)4 * H * IN * 2);
    __hip_bfloat16* whh1b = (__hip_bfloat16*)alloc((size_t)4 * H * H * 2);
    __hip_bfloat16* wih1b = (__hip_bfloat16*)alloc((size_t)4 * H * H * 2);
    __hip_bfloat16* wfcb  = (__hip_bfloat16*)alloc((size_t)OUT * H * 2);
    __hip_bfloat16* h0seq = (__hip_bfloat16*)alloc((size_t)T * B * H * 2);
    __hip_bfloat16* h1p0  = (__hip_bfloat16*)alloc((size_t)B * H * 2);
    __hip_bfloat16* h1p1  = (__hip_bfloat16*)alloc((size_t)B * H * 2);
    __hip_bfloat16* hzero = (__hip_bfloat16*)alloc((size_t)B * H * 2);
    float* c0 = (float*)alloc((size_t)B * H * 4);
    float* c1 = (float*)alloc((size_t)B * H * 4);

    auto cvt = [&](const float* src, __hip_bfloat16* dst, int n) {
        cvt_f32_bf16<<<(n + 255) / 256, 256, 0, stream>>>(src, dst, n);
    };
    cvt(x,    xbf,   B * T * IN);
    cvt(whh0, whh0b, 4 * H * H);
    cvt(wih0, wih0b, 4 * H * IN);
    cvt(whh1, whh1b, 4 * H * H);
    cvt(wih1, wih1b, 4 * H * H);
    cvt(wfc,  wfcb,  OUT * H);
    fill_zero_bf16<<<(B * H + 255) / 256, 256, 0, stream>>>(hzero, B * H);
    fill_zero_f32 <<<(B * H + 255) / 256, 256, 0, stream>>>(c0, B * H);
    fill_zero_f32 <<<(B * H + 255) / 256, 256, 0, stream>>>(c1, B * H);

    // Software-pipelined scan: launch i runs layer-0 step i (blocks 0..63)
    // and layer-1 step i-1 (blocks 64..127).  513 launches total.
    for (int i = 0; i <= T; ++i) {
        // ---- layer 0, step t0 = i (valid while i < T) ----
        int t0 = (i < T) ? i : (T - 1);
        const __hip_bfloat16* hp0 = (t0 == 0) ? hzero : (h0seq + (size_t)(t0 - 1) * B * H);
        __hip_bfloat16* hn0 = h0seq + (size_t)t0 * B * H;
        const __hip_bfloat16* x0 = xbf + (size_t)t0 * IN;
        int do0 = (i < T) ? 1 : 0;
        // ---- layer 1, step t1 = i-1 (valid while i >= 1) ----
        int t1 = (i >= 1) ? (i - 1) : 0;
        const __hip_bfloat16* hp1 = (t1 == 0) ? hzero : ((t1 & 1) ? h1p0 : h1p1);
        __hip_bfloat16* hn1 = (t1 & 1) ? h1p1 : h1p0;
        const __hip_bfloat16* x1 = h0seq + (size_t)t1 * B * H;
        int do1 = (i >= 1) ? 1 : 0;

        lstm_dual_step<<<128, 256, 0, stream>>>(
            hp0, x0, IN, T * IN, whh0b, wih0b, bih0, bhh0, c0, hn0, do0,
            hp1, x1, H,  H,      whh1b, wih1b, bih1, bhh1, c1, hn1, do1);
    }
    // Last layer-1 step is t = T-1 = 511 (odd) -> final h lives in h1p1.
    fc_kernel<<<4, 256, 0, stream>>>(h1p1, wfcb, bfc, out);
}
